// TtMoeLayer_1597727834772
// MI455X (gfx1250) — matrix operand order repaired and checked by reading, not compiled, not run on hardware
//
#include <hip/hip_runtime.h>
#include <math.h>

#define DI __device__ __forceinline__

typedef __attribute__((ext_vector_type(16))) __bf16 v16bf;
typedef __attribute__((ext_vector_type(8)))  __bf16 v8bf;
typedef __attribute__((ext_vector_type(2)))  __bf16 v2bf;
typedef __attribute__((ext_vector_type(8)))  float  v8f;

// ---- problem sizes (fixed by the reference) ----
constexpr int B_ = 2, S_ = 1024, H_ = 2048, F_ = 7168, E_ = 8, K_ = 2;
constexpr int T_ = B_ * S_;          // 2048 tokens
constexpr int NSLOT = T_ * K_;       // 4096 (token,k) slots

// ---- GEMM tiling: 128x128 block, 8 waves of 32x64, K-slab 32 ----
constexpr int BM = 128, BN = 128, BK = 32;
constexpr int AP = 40;               // LDS pitch A (32 + 8 pad, keeps 16B align)
constexpr int BP = 40;               // LDS pitch transposed B

// ---- workspace layout (bytes) ----
constexpr size_t OFF_CNT  = 0;                                 // E ints
constexpr size_t OFF_LIST = 256;                               // E*T ints
constexpr size_t OFF_WB   = OFF_LIST + (size_t)E_ * T_ * 4;    // NSLOT floats
constexpr size_t OFF_ZED  = OFF_WB   + (size_t)NSLOT * 4;      // 64B zero page
constexpr size_t OFF_XBF  = OFF_ZED  + 64;                     // T*H bf16
constexpr size_t OFF_HID  = OFF_XBF  + (size_t)T_ * H_ * 2;    // NSLOT*F bf16
constexpr size_t OFF_OSL  = OFF_HID  + (size_t)NSLOT * F_ * 2; // NSLOT*H f32

// ---- feature gates (compile-safe fallbacks) ----
#if defined(__has_builtin)
#if __has_builtin(__builtin_amdgcn_global_load_async_to_lds_b128) && \
    __has_builtin(__builtin_amdgcn_s_wait_asynccnt)
#define HAS_ASYNC_LDS 1
#endif
#if __has_builtin(__builtin_amdgcn_cvt_pk_bf16_f32)
#define HAS_CVT_PK 1
#endif
#endif
#ifndef HAS_ASYNC_LDS
#define HAS_ASYNC_LDS 0
#endif
#ifndef HAS_CVT_PK
#define HAS_CVT_PK 0
#endif

// ---- helpers ----
DI unsigned short bf_us(float f) {               // fp32 -> bf16 bits (RNE)
  unsigned u = __builtin_bit_cast(unsigned, f);
  u += 0x7FFFu + ((u >> 16) & 1u);
  return (unsigned short)(u >> 16);
}
DI __bf16 f2bf(float f) { return __builtin_bit_cast(__bf16, bf_us(f)); }

DI unsigned pack_bf16(float lo, float hi) {      // two bf16 in one dword
#if HAS_CVT_PK
  v2bf r = __builtin_amdgcn_cvt_pk_bf16_f32(lo, hi);
  return __builtin_bit_cast(unsigned, r);
#else
  return (unsigned)bf_us(lo) | ((unsigned)bf_us(hi) << 16);
#endif
}
DI void store_pk(__bf16* p, float lo, float hi) { *(unsigned*)p = pack_bf16(lo, hi); }

// 16B global -> LDS copy: CDNA5 async DMA when available, else through VGPRs
DI void cp16_g2l(__bf16* l, const __bf16* g) {
#if HAS_ASYNC_LDS
  typedef int v4i_ __attribute__((vector_size(16)));
  typedef __attribute__((address_space(1))) v4i_* gp_t;   // global (AS1) int4*
  typedef __attribute__((address_space(3))) v4i_* lp_t;   // LDS (AS3) int4*
  __builtin_amdgcn_global_load_async_to_lds_b128((gp_t)g, (lp_t)l, 0, 0);
#else
  *(uint4*)l = *(const uint4*)g;
#endif
}
DI void wait_async() {
#if HAS_ASYNC_LDS
  __builtin_amdgcn_s_wait_asynccnt(0);
#endif
}

// WMMA fragment from two 16B LDS runs (-> 2x ds_load_b128)
DI v16bf frag_ld(const __bf16* p0, const __bf16* p1) {
  v8bf lo = *(const v8bf*)p0;
  v8bf hi = *(const v8bf*)p1;
  return __builtin_shufflevector(lo, hi, 0,1,2,3,4,5,6,7,8,9,10,11,12,13,14,15);
}
DI v8f wmma_bf16(v16bf a, v16bf b, v8f c) {
  return __builtin_amdgcn_wmma_f32_16x16x32_bf16(false, a, false, b, (short)0, c, false, false);
}

// ================= kernel 0: zero counters + zero page, convert x -> bf16 =================
__global__ void k_init(const float* __restrict__ x, __bf16* __restrict__ xbf,
                       int* __restrict__ cnt, float* __restrict__ zed) {
  int i = blockIdx.x * blockDim.x + threadIdx.x;
  if (i < E_) cnt[i] = 0;
  if (i < 16) zed[i] = 0.0f;                    // 64B zero page for padded rows
  int i4 = i * 4;
  if (i4 < T_ * H_) {
    float4 v = *(const float4*)(x + i4);
    ushort4 o = make_ushort4(bf_us(v.x), bf_us(v.y), bf_us(v.z), bf_us(v.w));
    *(ushort4*)(xbf + i4) = o;
  }
}

// ================= kernel 1: gating (one wave32 per token) =================
__global__ void k_gate(const float* __restrict__ x, const float* __restrict__ gw,
                       int* __restrict__ cnt, int* __restrict__ list,
                       float* __restrict__ wb) {
  const int tid  = threadIdx.x;
  const int lane = tid & 31;
  const int t    = blockIdx.x * 8 + (tid >> 5);
  if (t >= T_) return;

  float acc[E_] = {};
  const float* xr = x + (size_t)t * H_;
  for (int h = lane; h < H_; h += 32) {
    float xv = xr[h];
    const float4* g = (const float4*)(gw + (size_t)h * E_);
    float4 g0 = g[0], g1 = g[1];
    acc[0] += xv * g0.x; acc[1] += xv * g0.y; acc[2] += xv * g0.z; acc[3] += xv * g0.w;
    acc[4] += xv * g1.x; acc[5] += xv * g1.y; acc[6] += xv * g1.z; acc[7] += xv * g1.w;
  }
#pragma unroll
  for (int off = 16; off > 0; off >>= 1)
#pragma unroll
    for (int e = 0; e < E_; ++e) acc[e] += __shfl_xor(acc[e], off, 32);

  if (lane == 0) {
    int i0 = 0; float v0 = acc[0];
    for (int e = 1; e < E_; ++e) if (acc[e] > v0) { v0 = acc[e]; i0 = e; } // ties -> lower idx
    int i1 = -1; float v1 = -3.4e38f;
    for (int e = 0; e < E_; ++e) if (e != i0 && acc[e] > v1) { v1 = acc[e]; i1 = e; }
    float e1 = __expf(v1 - v0);
    float s  = 1.0f + e1;
    wb[t * 2 + 0] = 1.0f / s;
    wb[t * 2 + 1] = e1 / s;
    int p0 = atomicAdd(&cnt[i0], 1); list[i0 * T_ + p0] = t * 2;
    int p1 = atomicAdd(&cnt[i1], 1); list[i1 * T_ + p1] = t * 2 + 1;
  }
}

// ===== kernel 2: grouped GEMM  hid = silu(Xg @ w1_e) * (Xg @ w3_e)  (bf16 WMMA) =====
__global__ __launch_bounds__(256) void k_mlp1(
    const __bf16* __restrict__ xbf, const float* __restrict__ w1,
    const float* __restrict__ w3, const int* __restrict__ cnt,
    const int* __restrict__ list, __bf16* __restrict__ hid,
    const __bf16* __restrict__ zbuf) {
  __shared__ __bf16 As[BM * AP];
  __shared__ __bf16 B1[BN * BP];   // transposed: [n][k]
  __shared__ __bf16 B3[BN * BP];

  const int e  = blockIdx.z;
  const int ce = cnt[e];
  const int mbase = blockIdx.y * BM;
  if (mbase >= ce) return;
  const int nbase = blockIdx.x * BN;

  const int tid = threadIdx.x;
  const int lane = tid & 31, wid = tid >> 5;
  const int wm = wid >> 1, wn = wid & 1;        // 4 x 2 waves -> each wave 32x64
  const int half = lane >> 4;
  const int mrow = lane & 15;

  // A loader: 2 threads per row, 16 bf16 (2 x 16B) each; padded rows read zero page
  const int arow = tid >> 1, acol = (tid & 1) * 16;
  const __bf16* aptr = nullptr;
  {
    int rl = mbase + arow;
    if (rl < ce) aptr = xbf + (size_t)(list[e * T_ + rl] >> 1) * H_;  // slot>>1 = token
  }
  // B loader: each thread owns K-row pair (kr,kr+1) x 8 cols -> packed dword LDS stores
  const int kr = (tid >> 4) * 2;                // 0..30 (even)
  const int cb = (tid & 15) * 8;                // 0..120
  const float* w1p = w1 + (size_t)e * H_ * F_ + (size_t)kr * F_ + nbase + cb;
  const float* w3p = w3 + (size_t)e * H_ * F_ + (size_t)kr * F_ + nbase + cb;

  v8f acc1[2][4] = {}, acc3[2][4] = {};

  for (int kk = 0; kk < H_; kk += BK) {
    __syncthreads();
    {   // A tile: async DMA overlaps with B conversion below
      const __bf16* g = aptr ? (aptr + kk + acol) : zbuf;
      __bf16* l = &As[arow * AP + acol];
      cp16_g2l(l, g);
      cp16_g2l(l + 8, g + 8);
    }
    {   // B tiles: fp32 -> bf16, packed pairs along K, transposed store
      const float* p1 = w1p + (size_t)kk * F_;
      const float* p3 = w3p + (size_t)kk * F_;
      if (kk + BK < H_) {
        __builtin_prefetch(p1 + (size_t)BK * F_, 0, 3);
        __builtin_prefetch(p1 + (size_t)(BK + 1) * F_, 0, 3);
        __builtin_prefetch(p3 + (size_t)BK * F_, 0, 3);
        __builtin_prefetch(p3 + (size_t)(BK + 1) * F_, 0, 3);
      }
      float4 a0 = *(const float4*)(p1);
      float4 a1 = *(const float4*)(p1 + 4);
      float4 c0 = *(const float4*)(p1 + F_);
      float4 c1 = *(const float4*)(p1 + F_ + 4);
      store_pk(&B1[(cb + 0) * BP + kr], a0.x, c0.x);
      store_pk(&B1[(cb + 1) * BP + kr], a0.y, c0.y);
      store_pk(&B1[(cb + 2) * BP + kr], a0.z, c0.z);
      store_pk(&B1[(cb + 3) * BP + kr], a0.w, c0.w);
      store_pk(&B1[(cb + 4) * BP + kr], a1.x, c1.x);
      store_pk(&B1[(cb + 5) * BP + kr], a1.y, c1.y);
      store_pk(&B1[(cb + 6) * BP + kr], a1.z, c1.z);
      store_pk(&B1[(cb + 7) * BP + kr], a1.w, c1.w);
      a0 = *(const float4*)(p3);
      a1 = *(const float4*)(p3 + 4);
      c0 = *(const float4*)(p3 + F_);
      c1 = *(const float4*)(p3 + F_ + 4);
      store_pk(&B3[(cb + 0) * BP + kr], a0.x, c0.x);
      store_pk(&B3[(cb + 1) * BP + kr], a0.y, c0.y);
      store_pk(&B3[(cb + 2) * BP + kr], a0.z, c0.z);
      store_pk(&B3[(cb + 3) * BP + kr], a0.w, c0.w);
      store_pk(&B3[(cb + 4) * BP + kr], a1.x, c1.x);
      store_pk(&B3[(cb + 5) * BP + kr], a1.y, c1.y);
      store_pk(&B3[(cb + 6) * BP + kr], a1.z, c1.z);
      store_pk(&B3[(cb + 7) * BP + kr], a1.w, c1.w);
    }
    wait_async();
    __syncthreads();

    v16bf af[2];
#pragma unroll
    for (int mi = 0; mi < 2; ++mi) {
      const __bf16* p = &As[(wm * 32 + mi * 16 + mrow) * AP + half * 16];
      af[mi] = frag_ld(p, p + 8);
    }
#pragma unroll
    for (int ni = 0; ni < 4; ++ni) {
      const __bf16* pb1 = &B1[(wn * 64 + ni * 16 + mrow) * BP + half * 16];
      const __bf16* pb3 = &B3[(wn * 64 + ni * 16 + mrow) * BP + half * 16];
      v16bf b1 = frag_ld(pb1, pb1 + 8);
      v16bf b3 = frag_ld(pb3, pb3 + 8);
#pragma unroll
      for (int mi = 0; mi < 2; ++mi) {
        acc1[mi][ni] = wmma_bf16(af[mi], b1, acc1[mi][ni]);
        acc3[mi][ni] = wmma_bf16(af[mi], b3, acc3[mi][ni]);
      }
    }
  }

  // fused SwiGLU + bf16 store (rows indexed by global slot id)
#pragma unroll
  for (int mi = 0; mi < 2; ++mi) {
#pragma unroll
    for (int r = 0; r < 8; ++r) {
      int rl = mbase + wm * 32 + mi * 16 + half * 8 + r;
      if (rl < ce) {
        int slot = list[e * T_ + rl];
        size_t rb = (size_t)slot * F_;
#pragma unroll
        for (int ni = 0; ni < 4; ++ni) {
          int col = nbase + wn * 64 + ni * 16 + mrow;
          float h1 = acc1[mi][ni][r];
          float h3 = acc3[mi][ni][r];
          float g  = (h1 / (1.0f + __expf(-h1))) * h3;   // silu(h1)*h3
          hid[rb + col] = f2bf(g);
        }
      }
    }
  }
}

// ===== kernel 3: grouped GEMM  out_slot = weight * (hid @ w2_e) =====
__global__ __launch_bounds__(256) void k_mlp2(
    const __bf16* __restrict__ hid, const float* __restrict__ w2,
    const int* __restrict__ cnt, const int* __restrict__ list,
    const float* __restrict__ wb, float* __restrict__ oslot,
    const __bf16* __restrict__ zbuf) {
  __shared__ __bf16 As[BM * AP];
  __shared__ __bf16 Bt[BN * BP];

  const int e  = blockIdx.z;
  const int ce = cnt[e];
  const int mbase = blockIdx.y * BM;
  if (mbase >= ce) return;
  const int nbase = blockIdx.x * BN;

  const int tid = threadIdx.x;
  const int lane = tid & 31, wid = tid >> 5;
  const int wm = wid >> 1, wn = wid & 1;
  const int half = lane >> 4;
  const int mrow = lane & 15;

  const int arow = tid >> 1, acol = (tid & 1) * 16;
  const __bf16* aptr = nullptr;
  {
    int rl = mbase + arow;
    if (rl < ce) aptr = hid + (size_t)list[e * T_ + rl] * F_;    // row = slot id
  }
  const int kr = (tid >> 4) * 2;
  const int cb = (tid & 15) * 8;
  const float* w2p = w2 + (size_t)e * F_ * H_ + (size_t)kr * H_ + nbase + cb;

  v8f acc[2][4] = {};

  for (int kk = 0; kk < F_; kk += BK) {
    __syncthreads();
    {
      const __bf16* g = aptr ? (aptr + kk + acol) : zbuf;
      __bf16* l = &As[arow * AP + acol];
      cp16_g2l(l, g);
      cp16_g2l(l + 8, g + 8);
    }
    {
      const float* p2 = w2p + (size_t)kk * H_;
      if (kk + BK < F_) {
        __builtin_prefetch(p2 + (size_t)BK * H_, 0, 3);
        __builtin_prefetch(p2 + (size_t)(BK + 1) * H_, 0, 3);
      }
      float4 a0 = *(const float4*)(p2);
      float4 a1 = *(const float4*)(p2 + 4);
      float4 c0 = *(const float4*)(p2 + H_);
      float4 c1 = *(const float4*)(p2 + H_ + 4);
      store_pk(&Bt[(cb + 0) * BP + kr], a0.x, c0.x);
      store_pk(&Bt[(cb + 1) * BP + kr], a0.y, c0.y);
      store_pk(&Bt[(cb + 2) * BP + kr], a0.z, c0.z);
      store_pk(&Bt[(cb + 3) * BP + kr], a0.w, c0.w);
      store_pk(&Bt[(cb + 4) * BP + kr], a1.x, c1.x);
      store_pk(&Bt[(cb + 5) * BP + kr], a1.y, c1.y);
      store_pk(&Bt[(cb + 6) * BP + kr], a1.z, c1.z);
      store_pk(&Bt[(cb + 7) * BP + kr], a1.w, c1.w);
    }
    wait_async();
    __syncthreads();

    v16bf af[2];
#pragma unroll
    for (int mi = 0; mi < 2; ++mi) {
      const __bf16* p = &As[(wm * 32 + mi * 16 + mrow) * AP + half * 16];
      af[mi] = frag_ld(p, p + 8);
    }
#pragma unroll
    for (int ni = 0; ni < 4; ++ni) {
      const __bf16* pb = &Bt[(wn * 64 + ni * 16 + mrow) * BP + half * 16];
      v16bf b = frag_ld(pb, pb + 8);
#pragma unroll
      for (int mi = 0; mi < 2; ++mi) acc[mi][ni] = wmma_bf16(af[mi], b, acc[mi][ni]);
    }
  }

#pragma unroll
  for (int mi = 0; mi < 2; ++mi) {
#pragma unroll
    for (int r = 0; r < 8; ++r) {
      int rl = mbase + wm * 32 + mi * 16 + half * 8 + r;
      if (rl < ce) {
        int slot = list[e * T_ + rl];
        float wgt = wb[slot];
        size_t rb = (size_t)slot * H_;
#pragma unroll
        for (int ni = 0; ni < 4; ++ni) {
          int col = nbase + wn * 64 + ni * 16 + mrow;
          oslot[rb + col] = wgt * acc[mi][ni][r];
        }
      }
    }
  }
}

// ================= kernel 4: combine the two slots per token =================
__global__ void k_combine(const float* __restrict__ oslot, float* __restrict__ out) {
  int i  = blockIdx.x * blockDim.x + threadIdx.x;
  int i4 = i * 4;
  if (i4 >= T_ * H_) return;
  int t = i4 >> 11;                 // H_ == 2048
  int h = i4 & (H_ - 1);
  float4 a = *(const float4*)(oslot + (size_t)(2 * t + 0) * H_ + h);
  float4 b = *(const float4*)(oslot + (size_t)(2 * t + 1) * H_ + h);
  float4 o = make_float4(a.x + b.x, a.y + b.y, a.z + b.z, a.w + b.w);
  *(float4*)(out + i4) = o;
}

extern "C" void kernel_launch(void* const* d_in, const int* in_sizes, int n_in,
                              void* d_out, int out_size, void* d_ws, size_t ws_size,
                              hipStream_t stream) {
  (void)in_sizes; (void)n_in; (void)out_size; (void)ws_size;
  const float* x  = (const float*)d_in[0];
  const float* gw = (const float*)d_in[1];
  const float* w1 = (const float*)d_in[2];
  const float* w3 = (const float*)d_in[3];
  const float* w2 = (const float*)d_in[4];
  float* out = (float*)d_out;

  char* ws = (char*)d_ws;
  int*    cnt  = (int*)(ws + OFF_CNT);
  int*    list = (int*)(ws + OFF_LIST);
  float*  wb   = (float*)(ws + OFF_WB);
  float*  zed  = (float*)(ws + OFF_ZED);
  __bf16* xbf  = (__bf16*)(ws + OFF_XBF);
  __bf16* hid  = (__bf16*)(ws + OFF_HID);
  float*  osl  = (float*)(ws + OFF_OSL);
  const __bf16* zbuf = (const __bf16*)(ws + OFF_ZED);

  const int n4 = T_ * H_ / 4;
  k_init<<<(n4 + 255) / 256, 256, 0, stream>>>(x, xbf, cnt, zed);
  k_gate<<<T_ / 8, 256, 0, stream>>>(x, gw, cnt, list, wb);
  k_mlp1<<<dim3(F_ / BN, T_ / BM, E_), 256, 0, stream>>>(xbf, w1, w3, cnt, list, hid, zbuf);
  k_mlp2<<<dim3(H_ / BN, T_ / BM, E_), 256, 0, stream>>>(hid, w2, cnt, list, wb, osl, zbuf);
  k_combine<<<(n4 + 255) / 256, 256, 0, stream>>>(osl, out);
}
